// Net_23106924052539
// MI455X (gfx1250) — matrix operand order repaired and checked by reading, not compiled, not run on hardware
//
#include <hip/hip_runtime.h>
#include <hip/hip_bf16.h>
#include <cstddef>

// ---------------------------------------------------------------------------
// 2-layer LSTM LM on MI455X (gfx1250, wave32, WMMA).
//   VOCAB=32000, EMB=HID=1024, B=16, T=256 -> M = B*T = 4096
// Pipeline:
//   1) weights f32 -> f16 (workspace)
//   2) embedding gather -> X (f16)
//   3) xproj = X @ w_ih^T + biases      (WMMA GEMM, f32 out)
//   4) 256 sequential lstm_step kernels per layer (WMMA per-step GEMM)
//   5) logits = H1 @ w_out^T + b_out    (WMMA GEMM, bandwidth-bound)
// K-loops are unrolled x2 with alternating fragment sets: rotation-free
// double buffering (no cross-iteration register copies), so each WMMA
// group's loads are issued ~half an iteration before they are consumed.
// ---------------------------------------------------------------------------

typedef __attribute__((ext_vector_type(16))) _Float16 v16h;
typedef __attribute__((ext_vector_type(8)))  _Float16 v8h;
typedef __attribute__((ext_vector_type(8)))  float    v8f;

union Frag16 { v16h v; v8h h[2]; };

// A-matrix fragment (16x32 f16, row-major A, leading dim lda).
// ISA layout: lanes 0-15 hold M=lane,     K = e<8 ? e   : e+8
//             lanes 16-31 hold M=lane-16, K = e<8 ? e+8 : e+16
__device__ __forceinline__ v16h load_a_frag(const _Float16* __restrict__ A,
                                            int lda, int row0, int k0, int lane) {
  int m  = lane & 15;
  int hi = lane >> 4;                        // 0 or 1
  const _Float16* p = A + (size_t)(row0 + m) * lda + k0 + hi * 8;
  Frag16 f;
  f.h[0] = *reinterpret_cast<const v8h*>(p);        // K = base .. base+7
  f.h[1] = *reinterpret_cast<const v8h*>(p + 16);   // K = base+16 .. base+23
  return f.v;
}

// B-matrix fragment (32x16 f16). Computing A @ W^T with W [N,K] row-major:
// B[k,n] = W[col0+n, k]; each lane's 16 K-values are two 8-wide runs, 16 apart, in one W row.
// ISA layout: lanes 0-15: N=lane, K = e<8 ? e : e+8 ; lanes 16-31: N=lane-16, K = e<8 ? e+8 : e+16
__device__ __forceinline__ v16h load_b_frag(const _Float16* __restrict__ W,
                                            int ldw, int col0, int k0, int lane) {
  int n  = lane & 15;
  int hi = lane >> 4;
  const _Float16* p = W + (size_t)(col0 + n) * ldw + k0 + hi * 8;
  Frag16 f;
  f.h[0] = *reinterpret_cast<const v8h*>(p);
  f.h[1] = *reinterpret_cast<const v8h*>(p + 16);
  return f.v;
}

// ---------------------------------------------------------------------------
// WMMA GEMM:  C[M,N] = A[M,K] @ W[N,K]^T + bias1[N] (+ bias2[N])
// 256 threads = 8 waves arranged 2(M) x 4(N); wave tile 16x64;
// block tile 32(M) x 256(N). All problem dims divide the tiles exactly;
// K is a multiple of 64.
// ---------------------------------------------------------------------------
__global__ __launch_bounds__(256) void gemm_f16_wmma(
    const _Float16* __restrict__ A, const _Float16* __restrict__ W,
    const float* __restrict__ bias1, const float* __restrict__ bias2,
    float* __restrict__ C, int M, int N, int K)
{
  const int lane = threadIdx.x & 31;
  const int wave = threadIdx.x >> 5;       // 0..7
  const int wm   = wave >> 2;              // 0..1
  const int wn   = wave & 3;               // 0..3
  const int row0 = blockIdx.y * 32 + wm * 16;
  const int col0 = blockIdx.x * 256 + wn * 64;

  v8f acc[4] = {};

  // Prologue: fragment set 0 holds K-step 0.
  v16h a0 = load_a_frag(A, K, row0, 0, lane);
  v16h b0_0 = load_b_frag(W, K, col0 +  0, 0, lane);
  v16h b0_1 = load_b_frag(W, K, col0 + 16, 0, lane);
  v16h b0_2 = load_b_frag(W, K, col0 + 32, 0, lane);
  v16h b0_3 = load_b_frag(W, K, col0 + 48, 0, lane);

  for (int k0 = 0; k0 < K; k0 += 64) {
    // Fragment set 1 <- K-step k0+32 (issued before consuming set 0).
    const int k1 = k0 + 32;
    v16h a1   = load_a_frag(A, K, row0, k1, lane);
    v16h b1_0 = load_b_frag(W, K, col0 +  0, k1, lane);
    v16h b1_1 = load_b_frag(W, K, col0 + 16, k1, lane);
    v16h b1_2 = load_b_frag(W, K, col0 + 32, k1, lane);
    v16h b1_3 = load_b_frag(W, K, col0 + 48, k1, lane);

    acc[0] = __builtin_amdgcn_wmma_f32_16x16x32_f16(false, a0, false, b0_0, (short)0, acc[0], false, false);
    acc[1] = __builtin_amdgcn_wmma_f32_16x16x32_f16(false, a0, false, b0_1, (short)0, acc[1], false, false);
    acc[2] = __builtin_amdgcn_wmma_f32_16x16x32_f16(false, a0, false, b0_2, (short)0, acc[2], false, false);
    acc[3] = __builtin_amdgcn_wmma_f32_16x16x32_f16(false, a0, false, b0_3, (short)0, acc[3], false, false);

    // Fragment set 0 <- K-step k0+64 (clamped on final trip; uniform).
    const int kp = (k0 + 64 < K) ? (k0 + 64) : 0;
    a0   = load_a_frag(A, K, row0, kp, lane);
    b0_0 = load_b_frag(W, K, col0 +  0, kp, lane);
    b0_1 = load_b_frag(W, K, col0 + 16, kp, lane);
    b0_2 = load_b_frag(W, K, col0 + 32, kp, lane);
    b0_3 = load_b_frag(W, K, col0 + 48, kp, lane);

    acc[0] = __builtin_amdgcn_wmma_f32_16x16x32_f16(false, a1, false, b1_0, (short)0, acc[0], false, false);
    acc[1] = __builtin_amdgcn_wmma_f32_16x16x32_f16(false, a1, false, b1_1, (short)0, acc[1], false, false);
    acc[2] = __builtin_amdgcn_wmma_f32_16x16x32_f16(false, a1, false, b1_2, (short)0, acc[2], false, false);
    acc[3] = __builtin_amdgcn_wmma_f32_16x16x32_f16(false, a1, false, b1_3, (short)0, acc[3], false, false);
  }

  // C/D layout: VGPR r -> M = row0 + r + (lane>=16)*8, N = col0 + (lane&15)
  const int mbase = row0 + (lane >> 4) * 8;
  const int nn    = lane & 15;
#pragma unroll
  for (int j = 0; j < 4; ++j) {
    const int col = col0 + j * 16 + nn;
    float bv = bias1 ? bias1[col] : 0.0f;
    if (bias2) bv += bias2[col];
#pragma unroll
    for (int r = 0; r < 8; ++r)
      C[(size_t)(mbase + r) * N + col] = acc[j][r] + bv;
  }
}

// ---------------------------------------------------------------------------
// One LSTM timestep: gates = h_in @ w_hh^T + xproj[:,t,:]; cell update.
// 64 blocks (16-column groups of H), 128 threads = 4 waves (i,f,g,o).
// ---------------------------------------------------------------------------
__global__ __launch_bounds__(128) void lstm_step(
    const _Float16* __restrict__ h_in,    // [16,1024] f16
    const _Float16* __restrict__ Whh,     // [4096,1024] f16
    const float*    __restrict__ xproj,   // [4096,4096] f32, row = b*T + t
    float*          __restrict__ c,       // [16,1024] f32 (cols disjoint per block)
    _Float16*       __restrict__ h_out,   // [16,1024] f16
    _Float16*       __restrict__ hs_out,  // [4096,1024] f16, row = b*T + t
    int t)
{
  __shared__ float gates[4][16][16];
  const int H = 1024, T = 256, G4 = 4096;
  const int lane = threadIdx.x & 31;
  const int g    = threadIdx.x >> 5;       // gate: 0=i 1=f 2=g 3=o
  const int col0 = blockIdx.x * 16;        // column group within H
  const int wcol = g * H + col0;           // row block in Whh

  v8f acc = {};
  v16h a0 = load_a_frag(h_in, H, 0, 0, lane);
  v16h b0 = load_b_frag(Whh, H, wcol, 0, lane);
  for (int k0 = 0; k0 < H; k0 += 64) {
    const int k1 = k0 + 32;
    v16h a1 = load_a_frag(h_in, H, 0, k1, lane);
    v16h b1 = load_b_frag(Whh, H, wcol, k1, lane);
    acc = __builtin_amdgcn_wmma_f32_16x16x32_f16(false, a0, false, b0, (short)0, acc, false, false);
    const int kp = (k0 + 64 < H) ? (k0 + 64) : 0;
    a0 = load_a_frag(h_in, H, 0, kp, lane);
    b0 = load_b_frag(Whh, H, wcol, kp, lane);
    acc = __builtin_amdgcn_wmma_f32_16x16x32_f16(false, a1, false, b1, (short)0, acc, false, false);
  }

  const int nn    = lane & 15;
  const int mbase = (lane >> 4) * 8;
#pragma unroll
  for (int r = 0; r < 8; ++r) {
    const int b = mbase + r;  // batch index
    gates[g][b][nn] = acc[r] + xproj[(size_t)(b * T + t) * G4 + g * H + col0 + nn];
  }
  __syncthreads();

  int idx = threadIdx.x * 2;
#pragma unroll
  for (int e = 0; e < 2; ++e, ++idx) {
    const int m = idx >> 4, n = idx & 15;
    const float iv = 1.0f / (1.0f + __expf(-gates[0][m][n]));
    const float fv = 1.0f / (1.0f + __expf(-gates[1][m][n]));
    const float gv = tanhf(gates[2][m][n]);
    const float ov = 1.0f / (1.0f + __expf(-gates[3][m][n]));
    const int col = col0 + n;
    const float cv = fv * c[m * H + col] + iv * gv;
    c[m * H + col] = cv;
    const float hv = ov * tanhf(cv);
    h_out[m * H + col] = (_Float16)hv;
    hs_out[(size_t)(m * T + t) * H + col] = (_Float16)hv;
  }
}

// ---------------------------------------------------------------------------
// Helpers: f32->f16 convert, embedding gather, state zeroing.
// ---------------------------------------------------------------------------
__global__ void cvt_f32_f16(const float* __restrict__ src,
                            _Float16* __restrict__ dst, size_t n) {
  size_t i = blockIdx.x * (size_t)blockDim.x + threadIdx.x;
  const size_t stride = gridDim.x * (size_t)blockDim.x;
  for (; i < n; i += stride) dst[i] = (_Float16)src[i];
}

__global__ void embed_gather(const int* __restrict__ tokens,
                             const float* __restrict__ emb,
                             _Float16* __restrict__ X) {
  const int row = blockIdx.x;                 // b*T + t, 0..4095
  const int tok = tokens[row];
  const float*    src = emb + (size_t)tok * 1024;
  _Float16*       dst = X   + (size_t)row * 1024;
  for (int i = threadIdx.x; i < 1024; i += blockDim.x)
    dst[i] = (_Float16)src[i];
}

__global__ void zero_state(_Float16* hA, _Float16* hB, float* c) {
  const int i = blockIdx.x * blockDim.x + threadIdx.x;
  if (i < 16 * 1024) {
    hA[i] = (_Float16)0.0f;
    hB[i] = (_Float16)0.0f;
    c[i]  = 0.0f;
  }
}

// ---------------------------------------------------------------------------
extern "C" void kernel_launch(void* const* d_in, const int* in_sizes, int n_in,
                              void* d_out, int out_size, void* d_ws, size_t ws_size,
                              hipStream_t stream) {
  (void)in_sizes; (void)n_in; (void)out_size; (void)ws_size;
  const int B = 16, T = 256, E = 1024, H = 1024, V = 32000;
  const int M = B * T;            // 4096
  const int G4 = 4 * H;           // 4096

  const int*   tokens = (const int*)  d_in[0];
  const float* emb    = (const float*)d_in[1];
  const float* w_ih0  = (const float*)d_in[2];
  const float* w_hh0  = (const float*)d_in[3];
  const float* b_ih0  = (const float*)d_in[4];
  const float* b_hh0  = (const float*)d_in[5];
  const float* w_ih1  = (const float*)d_in[6];
  const float* w_hh1  = (const float*)d_in[7];
  const float* b_ih1  = (const float*)d_in[8];
  const float* b_hh1  = (const float*)d_in[9];
  const float* w_out  = (const float*)d_in[10];
  const float* b_out  = (const float*)d_in[11];
  float* logits       = (float*)d_out;

  // Workspace carve-up (256B aligned slices).
  char* ws = (char*)d_ws;
  size_t off = 0;
  auto take = [&](size_t bytes) -> char* {
    char* p = ws + off;
    off += (bytes + 255) & ~(size_t)255;
    return p;
  };
  _Float16* wih0h = (_Float16*)take((size_t)G4 * E * 2);
  _Float16* whh0h = (_Float16*)take((size_t)G4 * H * 2);
  _Float16* wih1h = (_Float16*)take((size_t)G4 * H * 2);
  _Float16* whh1h = (_Float16*)take((size_t)G4 * H * 2);
  _Float16* wouth = (_Float16*)take((size_t)V  * H * 2);
  _Float16* X     = (_Float16*)take((size_t)M * E * 2);
  _Float16* H0    = (_Float16*)take((size_t)M * H * 2);
  _Float16* H1    = (_Float16*)take((size_t)M * H * 2);
  float*    XP    = (float*)   take((size_t)M * G4 * 4);
  _Float16* hA    = (_Float16*)take((size_t)B * H * 2);
  _Float16* hB    = (_Float16*)take((size_t)B * H * 2);
  float*    cS    = (float*)   take((size_t)B * H * 4);

  // 1) Weight conversion to f16.
  cvt_f32_f16<<<2048, 256, 0, stream>>>(w_ih0, wih0h, (size_t)G4 * E);
  cvt_f32_f16<<<2048, 256, 0, stream>>>(w_hh0, whh0h, (size_t)G4 * H);
  cvt_f32_f16<<<2048, 256, 0, stream>>>(w_ih1, wih1h, (size_t)G4 * H);
  cvt_f32_f16<<<2048, 256, 0, stream>>>(w_hh1, whh1h, (size_t)G4 * H);
  cvt_f32_f16<<<4096, 256, 0, stream>>>(w_out, wouth, (size_t)V * H);

  // 2) Embedding gather.
  embed_gather<<<M, 256, 0, stream>>>(tokens, emb, X);

  // 3) Layer 0 input projection: XP = X @ w_ih0^T + b_ih0 + b_hh0.
  gemm_f16_wmma<<<dim3(G4 / 256, M / 32), 256, 0, stream>>>(
      X, wih0h, b_ih0, b_hh0, XP, M, G4, E);

  // 4) Layer 0 recurrence (sequential, stream-ordered; ping-pong h).
  zero_state<<<64, 256, 0, stream>>>(hA, hB, cS);
  for (int t = 0; t < T; ++t) {
    const _Float16* hi = (t & 1) ? hB : hA;
    _Float16*       ho = (t & 1) ? hA : hB;
    lstm_step<<<64, 128, 0, stream>>>(hi, whh0h, XP, cS, ho, H0, t);
  }

  // 5) Layer 1 input projection: XP = H0 @ w_ih1^T + b_ih1 + b_hh1.
  gemm_f16_wmma<<<dim3(G4 / 256, M / 32), 256, 0, stream>>>(
      H0, wih1h, b_ih1, b_hh1, XP, M, G4, H);

  // 6) Layer 1 recurrence.
  zero_state<<<64, 256, 0, stream>>>(hA, hB, cS);
  for (int t = 0; t < T; ++t) {
    const _Float16* hi = (t & 1) ? hB : hA;
    _Float16*       ho = (t & 1) ? hA : hB;
    lstm_step<<<64, 128, 0, stream>>>(hi, whh1h, XP, cS, ho, H1, t);
  }

  // 7) Output projection: logits = H1 @ w_out^T + b_out (bandwidth-bound).
  gemm_f16_wmma<<<dim3(V / 256, M / 32), 256, 0, stream>>>(
      H1, wouth, b_out, nullptr, logits, M, V, H);
}
